// PNARandomEdgeUpdate_64811056496807
// MI455X (gfx1250) — compile-verified, matrix operand order checked by reading, not run on hardware
//
#include <hip/hip_runtime.h>
#include <math.h>

// ---------------------------------------------------------------------------
// PNA GNN forward for MI455X (gfx1250, wave32, WMMA + TDM).
// - All dense 128-out GEMMs: bf16 WMMA (v_wmma_f32_16x16x32_bf16), f32 accum.
// - Weights pre-converted to bf16 once, then DMA'd into LDS per K-chunk via
//   TENSOR_LOAD_TO_LDS (Tensor Data Mover, TENSORcnt).
// - Activation tiles staged via VALU (need f32->bf16 + PNA scaler fold-in).
// - Segment reductions via float global atomics (add/max_num/min_num).
// ---------------------------------------------------------------------------

typedef __attribute__((ext_vector_type(16))) __bf16 bf16x16;
typedef __attribute__((ext_vector_type(8)))  float  f32x8;
typedef __attribute__((ext_vector_type(4)))  unsigned int u32x4;
typedef __attribute__((ext_vector_type(8)))  int i32x8;
typedef __attribute__((ext_vector_type(4)))  int i32x4;

union AFrag { bf16x16 v; unsigned int u[8]; };

__device__ __forceinline__ unsigned int pack_bf16x2(float lo, float hi) {
  unsigned int ulo = __float_as_uint(lo) >> 16;
  unsigned int uhi = __float_as_uint(hi) & 0xffff0000u;
  return uhi | ulo;
}

// Issue a TDM load of a 128-row x 128-col bf16 tile (row stride ld_elems)
// from global memory into LDS at byte offset lds_off. D# per ISA 8.3/8.4.
__device__ __forceinline__ void tdm_load_tile(const unsigned short* gptr,
                                              unsigned int lds_off,
                                              int ld_elems) {
  unsigned long long ga = (unsigned long long)(size_t)gptr;
  u32x4 g0;
  g0[0] = 1u;                                   // count=1, user mode
  g0[1] = lds_off;                              // lds_addr
  g0[2] = (unsigned int)(ga & 0xffffffffu);     // global_addr[31:0]
  g0[3] = (unsigned int)((ga >> 32) & 0x1ffffffu) | (2u << 30);  // addr hi | type=2

  unsigned int td0 = (unsigned int)ld_elems;    // tensor_dim0 (row length)
  unsigned int td1 = 128u;                      // tensor_dim1 (rows)
  i32x8 g1 = {0, 0, 0, 0, 0, 0, 0, 0};
  g1[0] = (int)(1u << 16);                      // data_size=2B, no mask/pad/iter
  g1[1] = (int)((td0 & 0xffffu) << 16);         // tensor_dim0 low16 @ [63:48]
  g1[2] = (int)(((td0 >> 16) & 0xffffu) | ((td1 & 0xffffu) << 16)); // d0 hi | d1 lo
  g1[3] = (int)(((td1 >> 16) & 0xffffu) | (128u << 16));            // tile_dim0=128
  g1[4] = (int)(128u);                          // tile_dim1=128, tile_dim2=0
  g1[5] = (int)td0;                             // tensor_dim0_stride low32
  g1[6] = 0;                                    // stride hi16 | dim1_stride lo
  g1[7] = 0;

  i32x4 z4 = {0, 0, 0, 0};
  i32x8 z8 = {0, 0, 0, 0, 0, 0, 0, 0};
  // clang-23 / amdgpu-toolchain form: 6 args (extra i32x8 group before cpol)
  __builtin_amdgcn_tensor_load_to_lds(g0, g1, z4, z4, z8, 0);
}

enum {
  F_BIAS   = 1,
  F_RELU   = 2,
  F_TAIL   = 4,   // extra fp32 K-tail (RVD dims of init MLPs)
  F_GATHER = 8,   // += g1[src[row]] + g2[dst[row]]
  F_RESID  = 16,  // += (1+eps)*resid[row]
  F_POST2  = 32   // PNA scaler expansion: A is [N,512] agg, K=1536 virtual
};

// Output tile: 16 rows x 128 cols per block. 256 threads = 8 waves,
// wave w computes output columns [16w, 16w+16). K blocked by 128 via LDS.
// Weight chunk arrives via TDM; activation tile staged via VALU (bf16 cvt).
template<int F>
__global__ __launch_bounds__(256)
void gemm128(const unsigned short* __restrict__ Wbf,   // bf16 weights, tight
             const float* __restrict__ A, int lda,
             const float* __restrict__ W, int ldw,     // f32 weights (tail only)
             const float* __restrict__ bias,
             float* __restrict__ out,
             int M, int nChunks, int tail_len,
             const float* __restrict__ tailA, int ldtail,
             const int* __restrict__ srcI, const int* __restrict__ dstI,
             const float* __restrict__ g1, const float* __restrict__ g2,
             const float* __restrict__ resid, const float* __restrict__ epsp,
             const float* __restrict__ logd)
{
  __shared__ unsigned int sW[128 * 64];  // W chunk, bf16 pairs [n][k/2]   (32 KB)
  __shared__ unsigned int sA[16 * 64];   // A tile,  bf16 pairs [m][k/2]   (4 KB)
  __shared__ float sTA[16 * 12];         // fp32 K-tail of A
  __shared__ float sTW[128 * 12];        // fp32 K-tail of W

  const int tid  = threadIdx.x;
  const int wave = tid >> 5;
  const int lane = tid & 31;
  const int half = lane >> 4;
  const int l16  = lane & 15;
  const int row0 = blockIdx.x * 16;
  const int ldwbf = nChunks * 128;
  const unsigned int sWoff = (unsigned int)(size_t)&sW[0];  // LDS byte offset

  if (F & F_TAIL) {
    for (int i = tid; i < 16 * tail_len; i += 256) {
      int m = i / tail_len, j = i - m * tail_len;
      int row = row0 + m;
      sTA[m * 12 + j] = (row < M) ? tailA[(size_t)row * ldtail + j] : 0.f;
    }
    int km = nChunks * 128;
    for (int i = tid; i < 128 * tail_len; i += 256) {
      int n = i / tail_len, j = i - n * tail_len;
      sTW[n * 12 + j] = W[(size_t)n * ldw + km + j];
    }
  }

  f32x8 acc = {0.f, 0.f, 0.f, 0.f, 0.f, 0.f, 0.f, 0.f};

  for (int c = 0; c < nChunks; ++c) {
    __syncthreads();   // protect LDS vs previous chunk's readers
    const int k0  = (F & F_POST2) ? ((c & 3) * 128) : (c * 128);
    const int kw0 = c * 128;

    // Weight chunk (128x128 bf16, 32KB): Tensor Data Mover, wave 0 issues.
    if (wave == 0) {
      tdm_load_tile(Wbf + kw0, sWoff, ldwbf);
    }

    // Stage activation tile (16x128) as bf16 (PNA scaler folded in).
    for (int i = tid; i < 16 * 64; i += 256) {
      int m = i >> 6, p = i & 63;
      int row = row0 + m;
      float f0 = 0.f, f1 = 0.f;
      if (row < M) {
        const float* ap = A + (size_t)row * lda + k0 + 2 * p;
        f0 = ap[0]; f1 = ap[1];
        if (c + 1 < nChunks) __builtin_prefetch(ap + 128, 0, 1);
        if (F & F_POST2) {
          int grp = c >> 2;                 // 0:identity 1:amplify 2:attenuate
          float ld = logd[row];
          float sc = (grp == 0) ? 1.f : ((grp == 1) ? ld : 1.f / ld);
          f0 *= sc; f1 *= sc;
        }
      }
      sA[i] = pack_bf16x2(f0, f1);
    }

    if (wave == 0) __builtin_amdgcn_s_wait_tensorcnt(0);
    __syncthreads();

    #pragma unroll
    for (int ks = 0; ks < 4; ++ks) {
      AFrag a, b;
      #pragma unroll
      for (int v = 0; v < 8; ++v) {
        // 16-bit A-matrix 16x32 layout: lanes0-15 K=0..7,16..23; lanes16-31 +8
        int koff = ((v < 4) ? (2 * v) : (16 + 2 * (v - 4))) + 8 * half;
        int kp = (ks * 32 + koff) >> 1;
        a.u[v] = sA[l16 * 64 + kp];
        b.u[v] = sW[(wave * 16 + l16) * 64 + kp];  // B[k][n] = W[n][k]
      }
      acc = __builtin_amdgcn_wmma_f32_16x16x32_bf16(
          false, a.v, false, b.v, (short)0, acc, false, false);
    }
  }

  const int n = wave * 16 + l16;
  float bv = (F & F_BIAS) ? bias[n] : 0.f;
  float rs = 1.f;
  if (F & F_RESID) rs = 1.f + epsp[0];

  #pragma unroll
  for (int r = 0; r < 8; ++r) {
    int m = r + 8 * half;          // C layout: lanes16-31 hold M=8..15
    int row = row0 + m;
    if (row < M) {
      float v = acc[r] + bv;
      if (F & F_TAIL) {
        for (int j = 0; j < tail_len; ++j) v += sTA[m * 12 + j] * sTW[n * 12 + j];
      }
      if (F & F_GATHER) {
        int s = srcI[row], d = dstI[row];
        v += g1[(size_t)s * 128 + n] + g2[(size_t)d * 128 + n];
      }
      if (F & F_RELU) v = fmaxf(v, 0.f);
      if (F & F_RESID) v += rs * resid[(size_t)row * 128 + n];
      out[(size_t)row * 128 + n] = v;
    }
  }
}

// ------------------------- auxiliary kernels --------------------------------

// f32 weight (rows=128, ld_src cols) -> tight bf16 [128 x cols]
__global__ void conv_bf16(const float* __restrict__ s, int ld_src,
                          unsigned short* __restrict__ d, int cols, int total) {
  int i = blockIdx.x * blockDim.x + threadIdx.x;
  if (i < total) {
    int r = i / cols, c = i - r * cols;
    d[i] = (unsigned short)(__float_as_uint(s[(size_t)r * ld_src + c]) >> 16);
  }
}

__global__ void atom_embed(const int* __restrict__ feat, const float* __restrict__ emb,
                           float* __restrict__ out) {
  int row = blockIdx.x, col = threadIdx.x;
  __shared__ int f[9];
  if (col < 9) f[col] = feat[row * 9 + col];
  __syncthreads();
  float s = 0.f;
  #pragma unroll
  for (int c = 0; c < 9; ++c) s += emb[((size_t)c * 64 + f[c]) * 128 + col];
  out[(size_t)row * 128 + col] = s;
}

__global__ void bond_embed(const int* __restrict__ feat, const float* __restrict__ emb,
                           float* __restrict__ out) {
  int row = blockIdx.x, col = threadIdx.x;
  __shared__ int f[3];
  if (col < 3) f[col] = feat[row * 3 + col];
  __syncthreads();
  float s = 0.f;
  #pragma unroll
  for (int c = 0; c < 3; ++c) s += emb[((size_t)c * 8 + f[c]) * 128 + col];
  out[(size_t)row * 128 + col] = s;
}

__global__ void fill_kernel(float* __restrict__ p, float v, size_t n) {
  size_t i = (size_t)blockIdx.x * blockDim.x + threadIdx.x;
  if (i < n) p[i] = v;
}

__global__ void deg_scatter(const int* __restrict__ dst, float* __restrict__ deg, int E) {
  int i = blockIdx.x * blockDim.x + threadIdx.x;
  if (i < E) atomicAdd(&deg[dst[i]], 1.0f);
}

__global__ void logd_kernel(const float* __restrict__ deg, float* __restrict__ logd, int N) {
  int i = blockIdx.x * blockDim.x + threadIdx.x;
  if (i < N) {
    float d = deg[i];
    logd[i] = (d > 0.f) ? logf(d + 1.f) : 1.f;
  }
}

// agg layout per node: [0,128)=sum->mean  [128,256)=max  [256,384)=min  [384,512)=sumsq->std
__global__ void agg_init(float* __restrict__ agg, size_t total) {
  size_t i = (size_t)blockIdx.x * blockDim.x + threadIdx.x;
  if (i < total) {
    int col = (int)(i & 511);
    float v = 0.f;
    if (col >= 128 && col < 256) v = -3.0e38f;
    else if (col >= 256 && col < 384) v = 3.0e38f;
    agg[i] = v;
  }
}

__global__ void edge_scatter(const float* __restrict__ m, const int* __restrict__ dst,
                             float* __restrict__ agg) {
  int e = blockIdx.x, col = threadIdx.x;
  int d = dst[e];
  float v = m[(size_t)e * 128 + col];
  float* base = agg + (size_t)d * 512;
  atomicAdd(base + col, v);
  atomicMax(base + 128 + col, v);
  atomicMin(base + 256 + col, v);
  atomicAdd(base + 384 + col, v * v);
}

__global__ void agg_fin(float* __restrict__ agg, const float* __restrict__ deg) {
  int row = blockIdx.x, col = threadIdx.x;
  float d = deg[row];
  float ds = fmaxf(d, 1.f);
  float* base = agg + (size_t)row * 512;
  float s  = base[col];
  float sq = base[384 + col];
  float mean = s / ds;
  float msq  = sq / ds;
  float sd   = sqrtf(fmaxf(msq - mean * mean, 0.f) + 1e-5f);
  bool in = d > 0.f;
  base[col]       = in ? mean : 0.f;
  base[128 + col] = in ? base[128 + col] : 0.f;
  base[256 + col] = in ? base[256 + col] : 0.f;
  base[384 + col] = in ? sd : 0.f;
}

__global__ void gr_scatter(const float* __restrict__ h, const int* __restrict__ gid,
                           float* __restrict__ gs, float* __restrict__ gmax,
                           float* __restrict__ cnt) {
  int row = blockIdx.x, col = threadIdx.x;
  int g = gid[row];
  float v = h[(size_t)row * 128 + col];
  atomicAdd(&gs[(size_t)g * 128 + col], v);
  atomicMax(&gmax[(size_t)g * 128 + col], v);
  if (col == 0) atomicAdd(&cnt[g], 1.0f);
}

__global__ void gr_build(const float* __restrict__ gs, const float* __restrict__ gmax,
                         const float* __restrict__ cnt, float* __restrict__ ro) {
  int g = blockIdx.x, col = threadIdx.x;
  float c = cnt[g];
  float s = gs[(size_t)g * 128 + col];
  ro[(size_t)g * 384 + col]       = s;
  ro[(size_t)g * 384 + 128 + col] = s / fmaxf(c, 1.f);
  ro[(size_t)g * 384 + 256 + col] = (c > 0.f) ? gmax[(size_t)g * 128 + col] : 0.f;
}

// ------------------------------ driver --------------------------------------

extern "C" void kernel_launch(void* const* d_in, const int* in_sizes, int n_in,
                              void* d_out, int out_size, void* d_ws, size_t ws_size,
                              hipStream_t stream) {
  (void)in_sizes; (void)n_in; (void)out_size; (void)ws_size;
  const int N = 50000, E = 640000, NG = 100, DEPTH = 4;

  const int*   node_feat = (const int*)  d_in[0];
  const int*   edge_feat = (const int*)  d_in[1];
  const int*   src       = (const int*)  d_in[2];
  const int*   dst       = (const int*)  d_in[3];
  const int*   gid       = (const int*)  d_in[4];
  const float* rand_x    = (const float*)d_in[5];
  const float* rand_e    = (const float*)d_in[6];
  const float* atom_e    = (const float*)d_in[7];
  const float* bond_e    = (const float*)d_in[8];
  // params pytree (sorted dict keys): edge_init [9..12], layers [13..60],
  // node_init [61..64], out [65..68]
  #define PF(i) ((const float*)d_in[(i)])

  float* ws   = (float*)d_ws;
  float* h    = ws;                          // N*128
  float* hin  = h    + (size_t)N * 128;      // N*128
  float* hout = hin  + (size_t)N * 128;      // N*128
  float* e    = hout + (size_t)N * 128;      // E*128
  float* tmp  = e    + (size_t)E * 128;      // E*128
  float* agg  = tmp  + (size_t)E * 128;      // N*512
  float* deg  = agg  + (size_t)N * 512;      // N
  float* logd = deg  + N;                    // N
  float* gs   = logd + N;                    // NG*128
  float* gmax = gs   + NG * 128;             // NG*128
  float* cnt  = gmax + NG * 128;             // NG
  float* ro   = cnt  + 128;                  // NG*384
  float* mb   = ro   + NG * 384;             // NG*128

  // bf16 weight pool (converted once per launch)
  unsigned short* wb = (unsigned short*)(mb + (size_t)NG * 128);
  unsigned short* nbW0 = wb; wb += 128 * 128;
  unsigned short* nbW1 = wb; wb += 128 * 128;
  unsigned short* ebW0 = wb; wb += 128 * 128;
  unsigned short* ebW1 = wb; wb += 128 * 128;
  unsigned short* obW0 = wb; wb += 128 * 384;
  unsigned short* obW1 = wb; wb += 128 * 128;
  unsigned short* lWin[4], *lWout[4], *lWe[4], *lW1[4], *lWpre[4], *lW2[4];
  for (int l = 0; l < DEPTH; ++l) {
    lWin[l]  = wb; wb += 128 * 128;
    lWout[l] = wb; wb += 128 * 128;
    lWe[l]   = wb; wb += 128 * 128;
    lW1[l]   = wb; wb += 128 * 128;
    lWpre[l] = wb; wb += 128 * 128;
    lW2[l]   = wb; wb += 128 * 1536;
  }

  auto conv = [&](const float* s, int ld, unsigned short* d, int cols) {
    int total = 128 * cols;
    conv_bf16<<<(total + 255) / 256, 256, 0, stream>>>(s, ld, d, cols, total);
  };
  conv(PF(61), 138, nbW0, 128); conv(PF(63), 128, nbW1, 128);
  conv(PF(9),  138, ebW0, 128); conv(PF(11), 128, ebW1, 128);
  conv(PF(65), 384, obW0, 384); conv(PF(67), 128, obW1, 128);
  for (int l = 0; l < DEPTH; ++l) {
    const int b = 13 + 12 * l;
    conv(PF(b + 0), 128, lWin[l], 128);
    conv(PF(b + 1), 128, lWout[l], 128);
    conv(PF(b + 2), 128, lWe[l], 128);
    conv(PF(b + 6), 128, lW1[l], 128);
    conv(PF(b + 10), 128, lWpre[l], 128);
    conv(PF(b + 8), 1536, lW2[l], 1536);
  }

  const int NB = N / 16;                     // 3125
  const int EB = E / 16;                     // 40000
  const int GB = (NG + 15) / 16;             // 7

  // ---- embeddings -> h (atoms), e (bonds) ----
  atom_embed<<<N, 128, 0, stream>>>(node_feat, atom_e, h);
  bond_embed<<<E, 128, 0, stream>>>(edge_feat, bond_e, e);

  // ---- degree / log-degree ----
  fill_kernel<<<(N + 255) / 256, 256, 0, stream>>>(deg, 0.f, (size_t)N);
  deg_scatter<<<(E + 255) / 256, 256, 0, stream>>>(dst, deg, E);
  logd_kernel<<<(N + 255) / 256, 256, 0, stream>>>(deg, logd, N);

  // ---- node_init MLP: [h|rand_x](138) -> 128 relu -> 128 ----
  gemm128<F_BIAS | F_RELU | F_TAIL><<<NB, 256, 0, stream>>>(
      nbW0, h, 128, PF(61), 138, PF(62), hin, N, 1, 10, rand_x, 10,
      nullptr, nullptr, nullptr, nullptr, nullptr, nullptr, nullptr);
  gemm128<F_BIAS><<<NB, 256, 0, stream>>>(
      nbW1, hin, 128, nullptr, 128, PF(64), h, N, 1, 0, nullptr, 0,
      nullptr, nullptr, nullptr, nullptr, nullptr, nullptr, nullptr);

  // ---- edge_init MLP ----
  gemm128<F_BIAS | F_RELU | F_TAIL><<<EB, 256, 0, stream>>>(
      ebW0, e, 128, PF(9), 138, PF(10), tmp, E, 1, 10, rand_e, 10,
      nullptr, nullptr, nullptr, nullptr, nullptr, nullptr, nullptr);
  gemm128<F_BIAS><<<EB, 256, 0, stream>>>(
      ebW1, tmp, 128, nullptr, 128, PF(12), e, E, 1, 0, nullptr, 0,
      nullptr, nullptr, nullptr, nullptr, nullptr, nullptr, nullptr);

  // ---- message passing layers ----
  for (int l = 0; l < DEPTH; ++l) {
    const int b = 13 + 12 * l;

    // hin = h @ Win^T ; hout = h @ Wout^T
    gemm128<0><<<NB, 256, 0, stream>>>(
        lWin[l], h, 128, nullptr, 128, nullptr, hin, N, 1, 0, nullptr, 0,
        nullptr, nullptr, nullptr, nullptr, nullptr, nullptr, nullptr);
    gemm128<0><<<NB, 256, 0, stream>>>(
        lWout[l], h, 128, nullptr, 128, nullptr, hout, N, 1, 0, nullptr, 0,
        nullptr, nullptr, nullptr, nullptr, nullptr, nullptr, nullptr);

    // pre = relu(e @ We^T + b + hin[src] + hout[dst])
    gemm128<F_BIAS | F_RELU | F_GATHER><<<EB, 256, 0, stream>>>(
        lWe[l], e, 128, nullptr, 128, PF(b + 3), tmp, E, 1, 0, nullptr, 0,
        src, dst, hin, hout, nullptr, nullptr, nullptr);

    // e = (1+eps_e)*e + pre @ Wpre^T + b
    gemm128<F_BIAS | F_RESID><<<EB, 256, 0, stream>>>(
        lWpre[l], tmp, 128, nullptr, 128, PF(b + 11), e, E, 1, 0, nullptr, 0,
        nullptr, nullptr, nullptr, nullptr, e, PF(b + 4), nullptr);

    // m = e @ W1^T + b
    gemm128<F_BIAS><<<EB, 256, 0, stream>>>(
        lW1[l], e, 128, nullptr, 128, PF(b + 7), tmp, E, 1, 0, nullptr, 0,
        nullptr, nullptr, nullptr, nullptr, nullptr, nullptr, nullptr);

    // segment aggregation (sum/max/min/sumsq) then finalize to mean/max/min/std
    agg_init<<<(int)(((size_t)N * 512 + 255) / 256), 256, 0, stream>>>(agg, (size_t)N * 512);
    edge_scatter<<<E, 128, 0, stream>>>(tmp, dst, agg);
    agg_fin<<<N, 128, 0, stream>>>(agg, deg);

    // h = (1+eps_n)*h + scaled(agg) @ W2^T + b   (K=1536 virtual via POST2 mode)
    gemm128<F_BIAS | F_RESID | F_POST2><<<NB, 256, 0, stream>>>(
        lW2[l], agg, 512, nullptr, 1536, PF(b + 9), h, N, 12, 0, nullptr, 0,
        nullptr, nullptr, nullptr, nullptr, h, PF(b + 5), logd);
  }

  // ---- graph readout: sum / mean / max ----
  fill_kernel<<<(NG * 128 + 255) / 256, 256, 0, stream>>>(gs, 0.f, (size_t)NG * 128);
  fill_kernel<<<(NG * 128 + 255) / 256, 256, 0, stream>>>(gmax, -3.0e38f, (size_t)NG * 128);
  fill_kernel<<<1, 256, 0, stream>>>(cnt, 0.f, (size_t)NG);
  gr_scatter<<<N, 128, 0, stream>>>(h, gid, gs, gmax, cnt);
  gr_build<<<NG, 128, 0, stream>>>(gs, gmax, cnt, ro);

  // ---- output MLP: 384 -> 128 relu -> 128 ----
  gemm128<F_BIAS | F_RELU><<<GB, 256, 0, stream>>>(
      obW0, ro, 384, nullptr, 384, PF(66), mb, NG, 3, 0, nullptr, 0,
      nullptr, nullptr, nullptr, nullptr, nullptr, nullptr, nullptr);
  gemm128<F_BIAS><<<GB, 256, 0, stream>>>(
      obW1, mb, 128, nullptr, 128, PF(68), (float*)d_out, NG, 1, 0, nullptr, 0,
      nullptr, nullptr, nullptr, nullptr, nullptr, nullptr, nullptr);

  #undef PF
}
